// LSTM_DOUBLE_ATT_TOP_57990648430696
// MI455X (gfx1250) — matrix operand order repaired and checked by reading, not compiled, not run on hardware
//
#include <hip/hip_runtime.h>
#include <cstdint>
#include <cstddef>

// ---------------------------------------------------------------- types
typedef __bf16        v16bf __attribute__((ext_vector_type(16)));
typedef float         v8f   __attribute__((ext_vector_type(8)));
typedef unsigned int  u32x4 __attribute__((ext_vector_type(4)));
typedef unsigned int  u32x2 __attribute__((ext_vector_type(2)));
typedef float         f32x4 __attribute__((ext_vector_type(4)));

union Frag16 {               // one WMMA 16-bit operand fragment (8 VGPRs)
    v16bf v;
    u32x4 q[2];
};

// ---- explicit global-address-space accessors (force global_load/store, ----
// ---- never flat_*; POD ext_vector types so AS(1) deref is legal)       ----
#define AS_G __attribute__((address_space(1)))

__device__ __forceinline__ u32x4 gload4(const void* p) {
    return *(const AS_G u32x4*)(uintptr_t)p;
}
__device__ __forceinline__ u32x2 gload2(const void* p) {
    return *(const AS_G u32x2*)(uintptr_t)p;
}
__device__ __forceinline__ f32x4 gloadf4(const void* p) {
    return *(const AS_G f32x4*)(uintptr_t)p;
}
__device__ __forceinline__ float gloadf(const void* p) {
    return *(const AS_G float*)(uintptr_t)p;
}
__device__ __forceinline__ void gstore2(void* p, u32x2 v) {
    *(AS_G u32x2*)(uintptr_t)p = v;
}
__device__ __forceinline__ void gstoref(void* p, float v) {
    *(AS_G float*)(uintptr_t)p = v;
}
__device__ __forceinline__ void gstoreu16(void* p, unsigned short v) {
    *(AS_G unsigned short*)(uintptr_t)p = v;
}

__device__ __forceinline__ unsigned short f2bf(float f) {
    unsigned u = __float_as_uint(f);
    u += 0x7fffu + ((u >> 16) & 1u);           // round-to-nearest-even
    return (unsigned short)(u >> 16);
}
__device__ __forceinline__ float bf2f(unsigned b) {
    return __uint_as_float(b << 16);
}
__device__ __forceinline__ float sigm(float x) { return 1.0f / (1.0f + __expf(-x)); }

__device__ __forceinline__ v8f wmma_bf16(const Frag16& a, const Frag16& b, v8f c) {
    return __builtin_amdgcn_wmma_f32_16x16x32_bf16(
        /*neg_a=*/false, a.v, /*neg_b=*/false, b.v,
        /*c_mod=*/(short)0, c, /*reuse_a=*/false, /*reuse_b=*/false);
}

// ---------------------------------------------------------------- dims
#define BSZ   256
#define ATT_D 196
#define ATT_P 208        // 13 * 16
#define RNN_D 1024
#define G4    4096
#define MTOT  (BSZ * ATT_D)   // 50176 = 392 * 128

// ---------------------------------------------------------------- converts
// n4 = element count / 4 (all buffers are multiples of 4 elements)
__global__ __launch_bounds__(256) void conv_kernel(unsigned short* __restrict__ dst,
                                                   const float* __restrict__ src,
                                                   size_t n4) {
    size_t i = (size_t)blockIdx.x * blockDim.x + threadIdx.x;
    size_t stride = (size_t)gridDim.x * blockDim.x;
    for (; i < n4; i += stride) {
        f32x4 f = gloadf4(src + i * 4);
        u32x2 o;
        o[0] = (unsigned)f2bf(f[0]) | ((unsigned)f2bf(f[1]) << 16);
        o[1] = (unsigned)f2bf(f[2]) | ((unsigned)f2bf(f[3]) << 16);
        gstore2(dst + i * 4, o);
    }
}

// cols must be a multiple of 1024; rows beyond rows_src are zero-filled
__global__ __launch_bounds__(256) void convpad_kernel(unsigned short* __restrict__ dst,
                                                      const float* __restrict__ src,
                                                      int rows_src, int cols) {
    int r  = blockIdx.y;
    int c4 = (blockIdx.x * 256 + threadIdx.x) * 4;
    u32x2 o; o[0] = 0u; o[1] = 0u;
    if (r < rows_src) {
        f32x4 f = gloadf4(src + (size_t)r * cols + c4);
        o[0] = (unsigned)f2bf(f[0]) | ((unsigned)f2bf(f[1]) << 16);
        o[1] = (unsigned)f2bf(f[2]) | ((unsigned)f2bf(f[3]) << 16);
    }
    gstore2(dst + (size_t)r * cols + c4, o);
}

__global__ void padf_kernel(float* __restrict__ dst, const float* __restrict__ src,
                            int nvalid, int ntot) {
    int i = threadIdx.x;
    if (i < ntot) gstoref(dst + i, (i < nvalid) ? gloadf(src + i) : 0.0f);
}

// ---------------------------------------------------------------- generic WMMA GEMM
// out(MxN f32) = sum_p A_p(bf16 MxK, row-major) @ W_p(bf16 NpadxK, row-major)^T [+ bias]
// workgroup = 8 waves -> tile 32(M) x 256(N); M must be a multiple of 32.
__device__ __forceinline__ void gemm_accum(
    v8f acc[4],
    const unsigned short* __restrict__ A, const unsigned short* __restrict__ W,
    size_t rbase, size_t cbase, int K, int l16, int half)
{
    const unsigned short* arow  = A + (rbase + l16) * (size_t)K;
    const unsigned short* wbase = W + cbase * (size_t)K;
    for (int k0 = 0; k0 < K; k0 += 32) {
        Frag16 a;
        a.q[0] = gload4(arow + k0 + half * 16);
        a.q[1] = gload4(arow + k0 + half * 16 + 8);
#pragma unroll
        for (int nf = 0; nf < 4; ++nf) {
            const unsigned short* wr =
                wbase + (size_t)(nf * 16 + l16) * K + k0 + half * 16;
            Frag16 b;
            b.q[0] = gload4(wr);
            b.q[1] = gload4(wr + 8);
            acc[nf] = wmma_bf16(a, b, acc[nf]);
        }
    }
}

__global__ __launch_bounds__(256) void gemm_kernel(
    const unsigned short* __restrict__ A0, const unsigned short* __restrict__ W0,
    const unsigned short* __restrict__ A1, const unsigned short* __restrict__ W1,
    const unsigned short* __restrict__ A2, const unsigned short* __restrict__ W2,
    int K,
    const float* __restrict__ bias,
    float* __restrict__ out, int ldout, int Nvalid)
{
    const int tid  = threadIdx.x;
    const int wv   = tid >> 5;
    const int lane = tid & 31;
    const int half = lane >> 4;
    const int l16  = lane & 15;
    const int wrow = wv >> 2, wcol = wv & 3;

    const size_t rbase = (size_t)blockIdx.y * 32 + wrow * 16;
    const size_t cbase = (size_t)blockIdx.x * 256 + wcol * 64;

    v8f zero = {0.f, 0.f, 0.f, 0.f, 0.f, 0.f, 0.f, 0.f};
    v8f acc[4];
#pragma unroll
    for (int nf = 0; nf < 4; ++nf) acc[nf] = zero;

    gemm_accum(acc, A0, W0, rbase, cbase, K, l16, half);
    if (A1) gemm_accum(acc, A1, W1, rbase, cbase, K, l16, half);
    if (A2) gemm_accum(acc, A2, W2, rbase, cbase, K, l16, half);

#pragma unroll
    for (int nf = 0; nf < 4; ++nf) {
        int col = (int)cbase + nf * 16 + l16;
        if (col < Nvalid) {
            float bb = bias ? gloadf(bias + col) : 0.0f;
#pragma unroll
            for (int v = 0; v < 8; ++v) {
                size_t row = rbase + half * 8 + v;
                gstoref(out + row * (size_t)ldout + col, acc[nf][v] + bb);
            }
        }
    }
}

// ---------------------------------------------------------------- fused attention score
// For flattened rows g = b*196 + i of A = att_bf16 (MTOT x 1024):
//   dot[g] = sum_{j<196} tanh( (A@Wa^T)[g,j] + b_a2a[j] + att_h[g] ) * Wd[j] + b_d2d
// Wa is zero-padded to 208 rows, Wd/b_a2a zero-padded to 208 -> padded cols contribute 0.
// workgroup = 8 waves x 16 rows = 128 rows; each wave covers all 13 column tiles.
#define NT 13
__global__ __launch_bounds__(256) void att_score_kernel(
    const unsigned short* __restrict__ A,    // MTOT x 1024 bf16
    const unsigned short* __restrict__ Wp,   // 208 x 1024 bf16 (padded W_a2a)
    const float* __restrict__ bcol,          // 208 (padded b_a2a)
    const float* __restrict__ wd,            // 208 (padded W_d2d)
    const float* __restrict__ bd,            // scalar b_d2d
    const float* __restrict__ ath,           // MTOT (att_h flat)
    float* __restrict__ dot)                 // MTOT
{
    const int tid  = threadIdx.x;
    const int wv   = tid >> 5;
    const int lane = tid & 31;
    const int half = lane >> 4;
    const int l16  = lane & 15;

    const size_t rbase = (size_t)blockIdx.x * 128 + wv * 16;
    const unsigned short* arow = A + (rbase + l16) * (size_t)RNN_D;

    v8f zero = {0.f, 0.f, 0.f, 0.f, 0.f, 0.f, 0.f, 0.f};
    v8f acc[NT];
#pragma unroll
    for (int nt = 0; nt < NT; ++nt) acc[nt] = zero;

    for (int k0 = 0; k0 < RNN_D; k0 += 32) {
        Frag16 a;
        a.q[0] = gload4(arow + k0 + half * 16);
        a.q[1] = gload4(arow + k0 + half * 16 + 8);
#pragma unroll
        for (int nt = 0; nt < NT; ++nt) {
            const unsigned short* wr =
                Wp + (size_t)(nt * 16 + l16) * RNN_D + k0 + half * 16;
            Frag16 b;
            b.q[0] = gload4(wr);
            b.q[1] = gload4(wr + 8);
            acc[nt] = wmma_bf16(a, b, acc[nt]);
        }
    }

    // epilogue: tanh, scale by Wd, reduce over the 208 (13x16) columns
    float ah[8];
#pragma unroll
    for (int v = 0; v < 8; ++v) ah[v] = gloadf(ath + rbase + half * 8 + v);

    float s[8];
#pragma unroll
    for (int v = 0; v < 8; ++v) s[v] = 0.0f;

#pragma unroll
    for (int nt = 0; nt < NT; ++nt) {
        int col   = nt * 16 + l16;
        float wdc = gloadf(wd + col);
        float bc  = gloadf(bcol + col);
#pragma unroll
        for (int v = 0; v < 8; ++v)
            s[v] += wdc * tanhf(acc[nt][v] + bc + ah[v]);
    }

    float bdv = gloadf(bd);
#pragma unroll
    for (int v = 0; v < 8; ++v) {
        float t = s[v];
        t += __shfl_xor(t, 1, 32);
        t += __shfl_xor(t, 2, 32);
        t += __shfl_xor(t, 4, 32);
        t += __shfl_xor(t, 8, 32);   // reduced within each 16-lane half
        s[v] = t;
    }
    if (l16 == 0) {
#pragma unroll
        for (int v = 0; v < 8; ++v)
            gstoref(dot + rbase + half * 8 + v, s[v] + bdv);
    }
}

// ---------------------------------------------------------------- softmax over ATT per batch
__global__ __launch_bounds__(256) void softmax_kernel(const float* __restrict__ dot,
                                                      float* __restrict__ w) {
    __shared__ float red[256];
    const int b = blockIdx.x, t = threadIdx.x;
    float v = (t < ATT_D) ? gloadf(dot + (size_t)b * ATT_D + t) : -1e30f;
    red[t] = v;
    __syncthreads();
    for (int s = 128; s > 0; s >>= 1) {
        if (t < s) red[t] = fmaxf(red[t], red[t + s]);
        __syncthreads();
    }
    float mx = red[0];
    __syncthreads();
    float e = (t < ATT_D) ? __expf(v - mx) : 0.0f;
    red[t] = e;
    __syncthreads();
    for (int s = 128; s > 0; s >>= 1) {
        if (t < s) red[t] += red[t + s];
        __syncthreads();
    }
    float inv = 1.0f / red[0];
    if (t < ATT_D) gstoref(w + (size_t)b * ATT_D + t, e * inv);
}

// ---------------------------------------------------------------- weighted sum over ATT
// out[b,r] = sum_a w[b,a] * att[b,a,r]  (+ addend for top_h); one block per batch.
__global__ __launch_bounds__(256) void wsum_kernel(
    const unsigned short* __restrict__ attb,   // bf16 B x ATT x RNN
    const float* __restrict__ w,               // B x ATT
    unsigned short* __restrict__ out_bf,       // nullable: bf16 B x RNN
    float* __restrict__ out_f32,               // nullable: f32  B x RNN
    const float* __restrict__ addend)          // nullable (next_h for top_h)
{
    __shared__ float sw[ATT_D];
    const int b = blockIdx.x, tid = threadIdx.x;
    if (tid < ATT_D) sw[tid] = gloadf(w + (size_t)b * ATT_D + tid);
    __syncthreads();

    const unsigned short* base = attb + ((size_t)b * ATT_D) * RNN_D + tid * 4;
    float a0 = 0.f, a1 = 0.f, a2 = 0.f, a3 = 0.f;
    for (int a = 0; a < ATT_D; ++a) {
        u32x2 d = gload2(base + (size_t)a * RNN_D);
        float wa = sw[a];
        a0 += wa * bf2f(d[0] & 0xffffu);
        a1 += wa * bf2f(d[0] >> 16);
        a2 += wa * bf2f(d[1] & 0xffffu);
        a3 += wa * bf2f(d[1] >> 16);
    }
    size_t o = (size_t)b * RNN_D + tid * 4;
    if (out_bf) {
        u32x2 pk;
        pk[0] = (unsigned)f2bf(a0) | ((unsigned)f2bf(a1) << 16);
        pk[1] = (unsigned)f2bf(a2) | ((unsigned)f2bf(a3) << 16);
        gstore2(out_bf + o, pk);
    }
    if (out_f32) {
        gstoref(out_f32 + o + 0, a0 + gloadf(addend + o + 0));
        gstoref(out_f32 + o + 1, a1 + gloadf(addend + o + 1));
        gstoref(out_f32 + o + 2, a2 + gloadf(addend + o + 2));
        gstoref(out_f32 + o + 3, a3 + gloadf(addend + o + 3));
    }
}

// ---------------------------------------------------------------- LSTM gates
__global__ __launch_bounds__(256) void gates_kernel(
    const float* __restrict__ sums,            // B x 4096 (no bias yet)
    const float* __restrict__ b_i2h, const float* __restrict__ b_h2h,
    const float* __restrict__ b_a2h,
    const float* __restrict__ prev_c,
    float* __restrict__ next_c, float* __restrict__ next_h,
    unsigned short* __restrict__ nh_bf)
{
    int i = blockIdx.x * 256 + threadIdx.x;    // 0 .. B*RNN-1
    int b = i >> 10, r = i & 1023;
    const float* s = sums + (size_t)b * G4;
    float si = gloadf(s + r)        + gloadf(b_i2h + r)        + gloadf(b_h2h + r)        + gloadf(b_a2h + r);
    float sf = gloadf(s + 1024 + r) + gloadf(b_i2h + 1024 + r) + gloadf(b_h2h + 1024 + r) + gloadf(b_a2h + 1024 + r);
    float so = gloadf(s + 2048 + r) + gloadf(b_i2h + 2048 + r) + gloadf(b_h2h + 2048 + r) + gloadf(b_a2h + 2048 + r);
    float sg = gloadf(s + 3072 + r) + gloadf(b_i2h + 3072 + r) + gloadf(b_h2h + 3072 + r) + gloadf(b_a2h + 3072 + r);
    float ig = sigm(si), fg = sigm(sf), og = sigm(so), gt = tanhf(sg);
    float nc = fg * gloadf(prev_c + i) + ig * gt;
    float nh = og * tanhf(nc);
    gstoref(next_c + i, nc);
    gstoref(next_h + i, nh);
    gstoreu16(nh_bf + i, f2bf(nh));
}

// ---------------------------------------------------------------- workspace layout (bytes)
static constexpr size_t SZ_ATT_BF = (size_t)MTOT * RNN_D * 2;        // 102,760,448
static constexpr size_t SZ_VEC_BF = (size_t)BSZ * RNN_D * 2;         // 524,288
static constexpr size_t SZ_W208   = (size_t)ATT_P * RNN_D * 2;       // 425,984
static constexpr size_t SZ_W256   = (size_t)256 * RNN_D * 2;         // 524,288
static constexpr size_t SZ_WBIG   = (size_t)G4 * RNN_D * 2;          // 8,388,608
static constexpr size_t SZ_VSMALL = 1024;                            // padded 208-f32 vecs
static constexpr size_t SZ_MVEC   = (size_t)MTOT * 4;                // 200,704

static constexpr size_t OFF_ATT_BF  = 0;
static constexpr size_t OFF_X_BF    = OFF_ATT_BF  + SZ_ATT_BF;
static constexpr size_t OFF_H_BF    = OFF_X_BF    + SZ_VEC_BF;
static constexpr size_t OFF_ARES_BF = OFF_H_BF    + SZ_VEC_BF;
static constexpr size_t OFF_NH_BF   = OFF_ARES_BF + SZ_VEC_BF;
static constexpr size_t OFF_WA2A_P  = OFF_NH_BF   + SZ_VEC_BF;
static constexpr size_t OFF_WA2A1_P = OFF_WA2A_P  + SZ_W208;
static constexpr size_t OFF_WH2A_P  = OFF_WA2A1_P + SZ_W208;
static constexpr size_t OFF_WH2A1_P = OFF_WH2A_P  + SZ_W256;
static constexpr size_t OFF_WI2H    = OFF_WH2A1_P + SZ_W256;
static constexpr size_t OFF_WH2H    = OFF_WI2H    + SZ_WBIG;
static constexpr size_t OFF_WA2H    = OFF_WH2H    + SZ_WBIG;
static constexpr size_t OFF_WD_P    = OFF_WA2H    + SZ_WBIG;
static constexpr size_t OFF_WD1_P   = OFF_WD_P    + SZ_VSMALL;
static constexpr size_t OFF_BA2A_P  = OFF_WD1_P   + SZ_VSMALL;
static constexpr size_t OFF_BA2A1_P = OFF_BA2A_P  + SZ_VSMALL;
static constexpr size_t OFF_ATTH    = OFF_BA2A1_P + SZ_VSMALL;
static constexpr size_t OFF_ATTH1   = OFF_ATTH    + SZ_MVEC;
static constexpr size_t OFF_DOT0    = OFF_ATTH1   + SZ_MVEC;
static constexpr size_t OFF_DOT1    = OFF_DOT0    + SZ_MVEC;
static constexpr size_t OFF_SMW0    = OFF_DOT1    + SZ_MVEC;
static constexpr size_t OFF_SMW1    = OFF_SMW0    + SZ_MVEC;
static constexpr size_t OFF_SUMS    = OFF_SMW1    + SZ_MVEC;

extern "C" void kernel_launch(void* const* d_in, const int* in_sizes, int n_in,
                              void* d_out, int out_size, void* d_ws, size_t ws_size,
                              hipStream_t stream) {
    const float* x      = (const float*)d_in[0];
    const float* att    = (const float*)d_in[1];
    const float* prev_h = (const float*)d_in[2];
    const float* prev_c = (const float*)d_in[3];
    const float* W_a2a  = (const float*)d_in[4];
    const float* b_a2a  = (const float*)d_in[5];
    const float* W_h2a  = (const float*)d_in[6];
    const float* b_h2a  = (const float*)d_in[7];
    const float* W_d2d  = (const float*)d_in[8];
    const float* b_d2d  = (const float*)d_in[9];
    const float* W_a2h  = (const float*)d_in[10];
    const float* b_a2h  = (const float*)d_in[11];
    const float* W_i2h  = (const float*)d_in[12];
    const float* b_i2h  = (const float*)d_in[13];
    const float* W_h2h  = (const float*)d_in[14];
    const float* b_h2h  = (const float*)d_in[15];
    const float* W_a2a1 = (const float*)d_in[16];
    const float* b_a2a1 = (const float*)d_in[17];
    const float* W_h2a1 = (const float*)d_in[18];
    const float* b_h2a1 = (const float*)d_in[19];
    const float* W_d2d1 = (const float*)d_in[20];
    const float* b_d2d1 = (const float*)d_in[21];

    char* w8 = (char*)d_ws;
    unsigned short* att_bf  = (unsigned short*)(w8 + OFF_ATT_BF);
    unsigned short* x_bf    = (unsigned short*)(w8 + OFF_X_BF);
    unsigned short* h_bf    = (unsigned short*)(w8 + OFF_H_BF);
    unsigned short* ares_bf = (unsigned short*)(w8 + OFF_ARES_BF);
    unsigned short* nh_bf   = (unsigned short*)(w8 + OFF_NH_BF);
    unsigned short* wa2a_p  = (unsigned short*)(w8 + OFF_WA2A_P);
    unsigned short* wa2a1_p = (unsigned short*)(w8 + OFF_WA2A1_P);
    unsigned short* wh2a_p  = (unsigned short*)(w8 + OFF_WH2A_P);
    unsigned short* wh2a1_p = (unsigned short*)(w8 + OFF_WH2A1_P);
    unsigned short* wi2h_bf = (unsigned short*)(w8 + OFF_WI2H);
    unsigned short* wh2h_bf = (unsigned short*)(w8 + OFF_WH2H);
    unsigned short* wa2h_bf = (unsigned short*)(w8 + OFF_WA2H);
    float* wd_p    = (float*)(w8 + OFF_WD_P);
    float* wd1_p   = (float*)(w8 + OFF_WD1_P);
    float* ba2a_p  = (float*)(w8 + OFF_BA2A_P);
    float* ba2a1_p = (float*)(w8 + OFF_BA2A1_P);
    float* atth    = (float*)(w8 + OFF_ATTH);
    float* atth1   = (float*)(w8 + OFF_ATTH1);
    float* dot0    = (float*)(w8 + OFF_DOT0);
    float* dot1    = (float*)(w8 + OFF_DOT1);
    float* smw0    = (float*)(w8 + OFF_SMW0);
    float* smw1    = (float*)(w8 + OFF_SMW1);
    float* sums    = (float*)(w8 + OFF_SUMS);

    float* out_c   = (float*)d_out;
    float* out_h   = out_c + (size_t)BSZ * RNN_D;
    float* out_top = out_h + (size_t)BSZ * RNN_D;

    // ---- precision conversion / padding --------------------------------
    conv_kernel<<<8192, 256, 0, stream>>>(att_bf, att, (size_t)MTOT * RNN_D / 4);
    conv_kernel<<<256, 256, 0, stream>>>(x_bf, x, (size_t)BSZ * RNN_D / 4);
    conv_kernel<<<256, 256, 0, stream>>>(h_bf, prev_h, (size_t)BSZ * RNN_D / 4);
    convpad_kernel<<<dim3(1, ATT_P), 256, 0, stream>>>(wa2a_p,  W_a2a,  ATT_D, RNN_D);
    convpad_kernel<<<dim3(1, ATT_P), 256, 0, stream>>>(wa2a1_p, W_a2a1, ATT_D, RNN_D);
    convpad_kernel<<<dim3(1, 256),   256, 0, stream>>>(wh2a_p,  W_h2a,  ATT_D, RNN_D);
    convpad_kernel<<<dim3(1, 256),   256, 0, stream>>>(wh2a1_p, W_h2a1, ATT_D, RNN_D);
    convpad_kernel<<<dim3(1, G4),    256, 0, stream>>>(wi2h_bf, W_i2h,  G4, RNN_D);
    convpad_kernel<<<dim3(1, G4),    256, 0, stream>>>(wh2h_bf, W_h2h,  G4, RNN_D);
    convpad_kernel<<<dim3(1, G4),    256, 0, stream>>>(wa2h_bf, W_a2h,  G4, RNN_D);
    padf_kernel<<<1, 256, 0, stream>>>(wd_p,    W_d2d,  ATT_D, ATT_P);
    padf_kernel<<<1, 256, 0, stream>>>(wd1_p,   W_d2d1, ATT_D, ATT_P);
    padf_kernel<<<1, 256, 0, stream>>>(ba2a_p,  b_a2a,  ATT_D, ATT_P);
    padf_kernel<<<1, 256, 0, stream>>>(ba2a1_p, b_a2a1, ATT_D, ATT_P);

    // ---- attend #1 (prev_h) --------------------------------------------
    gemm_kernel<<<dim3(1, 8), 256, 0, stream>>>(
        h_bf, wh2a_p, nullptr, nullptr, nullptr, nullptr,
        RNN_D, b_h2a, atth, ATT_D, ATT_D);
    att_score_kernel<<<MTOT / 128, 256, 0, stream>>>(
        att_bf, wa2a_p, ba2a_p, wd_p, b_d2d, atth, dot0);
    softmax_kernel<<<BSZ, 256, 0, stream>>>(dot0, smw0);
    wsum_kernel<<<BSZ, 256, 0, stream>>>(att_bf, smw0, ares_bf, nullptr, nullptr);

    // ---- LSTM core ------------------------------------------------------
    gemm_kernel<<<dim3(16, 8), 256, 0, stream>>>(
        x_bf, wi2h_bf, h_bf, wh2h_bf, ares_bf, wa2h_bf,
        RNN_D, nullptr, sums, G4, G4);
    gates_kernel<<<(BSZ * RNN_D) / 256, 256, 0, stream>>>(
        sums, b_i2h, b_h2h, b_a2h, prev_c, out_c, out_h, nh_bf);

    // ---- attend #2 (next_h) + top_h ------------------------------------
    gemm_kernel<<<dim3(1, 8), 256, 0, stream>>>(
        nh_bf, wh2a1_p, nullptr, nullptr, nullptr, nullptr,
        RNN_D, b_h2a1, atth1, ATT_D, ATT_D);
    att_score_kernel<<<MTOT / 128, 256, 0, stream>>>(
        att_bf, wa2a1_p, ba2a1_p, wd1_p, b_d2d1, atth1, dot1);
    softmax_kernel<<<BSZ, 256, 0, stream>>>(dot1, smw1);
    wsum_kernel<<<BSZ, 256, 0, stream>>>(att_bf, smw1, nullptr, out_top, out_h);

    (void)in_sizes; (void)n_in; (void)out_size; (void)ws_size;
}
